// Encoder_81647328297152
// MI455X (gfx1250) — matrix operand order repaired and checked by reading, not compiled, not run on hardware
//
#include <hip/hip_runtime.h>
#include <hip/hip_bf16.h>

typedef __attribute__((ext_vector_type(16))) _Float16 v16h;
typedef __attribute__((ext_vector_type(8)))  _Float16 v8h;
typedef __attribute__((ext_vector_type(4)))  _Float16 v4h;
typedef __attribute__((ext_vector_type(8)))  float    v8f;
typedef __attribute__((ext_vector_type(4)))  unsigned int v4u;
typedef __attribute__((ext_vector_type(4)))  int      v4i;
typedef __attribute__((ext_vector_type(8)))  int      v8i;

#define NB   4
#define TSEQ 1024
#define CDIM 1024
#define NH   16
#define HDIM 64
#define NL   4
#define FFD  100
#define FFP  128   // FF padded to WMMA-friendly multiple

#if __has_builtin(__builtin_amdgcn_tensor_load_to_lds) && __has_builtin(__builtin_amdgcn_s_wait_tensorcnt)
#define HAVE_TDM 1
#else
#define HAVE_TDM 0
#endif

#if HAVE_TDM
// Issue one Tensor-Data-Mover 2D tile load (D# per CDNA5 ISA ch.8):
// rows x cols f16 tile from global (row stride = ld elems) into LDS,
// with LDS row padding expressed via pad_interval/pad_amount codes.
__device__ __forceinline__ void tdm_load_2d(const void* gsrc, void* ldst,
                                            int rows, int cols, int ld,
                                            int padi, int pada) {
  uint64_t ga = (uint64_t)(uintptr_t)gsrc;
  uint32_t la = (uint32_t)(uintptr_t)ldst;          // low 32 bits = LDS offset
  // Group0: count=1 | lds_addr | global_addr[56:0] | type=2
  v4u g0 = { 1u, la, (unsigned)(ga & 0xffffffffu),
             (unsigned)((ga >> 32) & 0x01ffffffu) | (2u << 30) };
  // Group1
  unsigned d0 = (1u << 16)                      // data_size = 2 bytes
              | (1u << 20)                      // pad_enable
              | ((unsigned)padi << 22)          // pad_interval code
              | ((unsigned)pada << 25);         // pad_amount code
  unsigned d1 = ((unsigned)cols & 0xffffu) << 16;                       // tensor_dim0 lo
  unsigned d2 = (((unsigned)cols >> 16) & 0xffffu)
              | (((unsigned)rows & 0xffffu) << 16);                     // td0 hi | tensor_dim1 lo
  unsigned d3 = (((unsigned)rows >> 16) & 0xffffu)
              | (((unsigned)cols & 0xffffu) << 16);                     // td1 hi | tile_dim0
  unsigned d4 = (unsigned)rows & 0xffffu;                               // tile_dim1
  unsigned d5 = (unsigned)ld;                                           // tensor_dim0_stride lo
  v8i g1 = { (int)d0, (int)d1, (int)d2, (int)d3, (int)d4, (int)d5, 0, 0 };
  v4i z4 = { 0, 0, 0, 0 };
#if __clang_major__ >= 23
  v8i z8 = { 0, 0, 0, 0, 0, 0, 0, 0 };
  __builtin_amdgcn_tensor_load_to_lds(g0, g1, z4, z4, z8, 0);
#else
  __builtin_amdgcn_tensor_load_to_lds(g0, g1, z4, z4, 0);
#endif
}
#endif

__device__ __forceinline__ void wait_stage() {
#if HAVE_TDM
  __builtin_amdgcn_s_wait_tensorcnt(0);
#endif
}

// ---------------------------------------------------------------- utilities
__global__ void zero4_kernel(float* p, size_t n4) {
  size_t i = (size_t)blockIdx.x * 256 + threadIdx.x;
  if (i < n4) ((float4*)p)[i] = make_float4(0.f, 0.f, 0.f, 0.f);
}

__global__ void cvt4_f32_f16_kernel(const float* __restrict__ src, _Float16* __restrict__ dst, size_t n4) {
  size_t i = (size_t)blockIdx.x * 256 + threadIdx.x;
  if (i >= n4) return;
  float4 v = ((const float4*)src)[i];
  v4h o = {(_Float16)v.x, (_Float16)v.y, (_Float16)v.z, (_Float16)v.w};
  ((v4h*)dst)[i] = o;
}

// Tiled transpose + f32->f16 + zero-pad: dst[Np][Kp] = src[K][N]^T
__global__ __launch_bounds__(256)
void cvtT_kernel(const float* __restrict__ src, _Float16* __restrict__ dst,
                 int K, int N, int Kp, int Np) {
  __shared__ float tile[32][33];
  int tk0 = blockIdx.x * 32, tn0 = blockIdx.y * 32;
  int tx = threadIdx.x & 31, ty = threadIdx.x >> 5;
  for (int r = ty; r < 32; r += 8) {
    int k = tk0 + r, n = tn0 + tx;
    tile[r][tx] = (k < K && n < N) ? src[(size_t)k * N + n] : 0.f;
  }
  __syncthreads();
  for (int r = ty; r < 32; r += 8) {
    int n = tn0 + r, k = tk0 + tx;
    if (n < Np && k < Kp) dst[(size_t)n * Kp + k] = (_Float16)tile[tx][r];
  }
}

__global__ void embed_kernel(const int* __restrict__ idx, const float* __restrict__ tok,
                             const float* __restrict__ pos, float* __restrict__ x) {
  size_t i = (size_t)blockIdx.x * 256 + threadIdx.x;   // over B*T*C/4
  if (i >= (size_t)NB * TSEQ * CDIM / 4) return;
  int c4 = (int)(i & (CDIM / 4 - 1));
  int bt = (int)(i >> 8);
  int t  = bt & (TSEQ - 1);
  float4 tv = ((const float4*)(tok + (size_t)idx[bt] * CDIM))[c4];
  float4 pv = ((const float4*)(pos + (size_t)t * CDIM))[c4];
  ((float4*)x)[i] = make_float4(tv.x + pv.x, tv.y + pv.y, tv.z + pv.z, tv.w + pv.w);
}

// ---------------------------------------------------------------- layernorm
__global__ __launch_bounds__(256)
void ln_kernel(const float* __restrict__ x, const float* __restrict__ w,
               const float* __restrict__ b, _Float16* out16, float* out32) {
  __shared__ float red[8];
  int row = blockIdx.x, tid = threadIdx.x;
  const float4 v = ((const float4*)(x + (size_t)row * CDIM))[tid];

  float s = v.x + v.y + v.z + v.w;
  for (int off = 16; off; off >>= 1) s += __shfl_down(s, off, 32);
  if ((tid & 31) == 0) red[tid >> 5] = s;
  __syncthreads();
  if (tid == 0) { float t = 0.f; for (int i = 0; i < 8; ++i) t += red[i]; red[0] = t * (1.f / CDIM); }
  __syncthreads();
  float mean = red[0];
  __syncthreads();

  float dx = v.x - mean, dy = v.y - mean, dz = v.z - mean, dw = v.w - mean;
  float var = dx * dx + dy * dy + dz * dz + dw * dw;
  for (int off = 16; off; off >>= 1) var += __shfl_down(var, off, 32);
  if ((tid & 31) == 0) red[tid >> 5] = var;
  __syncthreads();
  if (tid == 0) { float t = 0.f; for (int i = 0; i < 8; ++i) t += red[i]; red[0] = rsqrtf(t * (1.f / CDIM) + 1e-5f); }
  __syncthreads();
  float rstd = red[0];

  float4 wv = ((const float4*)w)[tid];
  float4 bv = ((const float4*)b)[tid];
  float o0 = dx * rstd * wv.x + bv.x;
  float o1 = dy * rstd * wv.y + bv.y;
  float o2 = dz * rstd * wv.z + bv.z;
  float o3 = dw * rstd * wv.w + bv.w;
  if (out16) {
    v4h o = {(_Float16)o0, (_Float16)o1, (_Float16)o2, (_Float16)o3};
    ((v4h*)(out16 + (size_t)row * CDIM))[tid] = o;
  }
  if (out32) ((float4*)(out32 + (size_t)row * CDIM))[tid] = make_float4(o0, o1, o2, o3);
}

// ---------------------------------------------------------------- WMMA GEMM
// out[M,N] = A[M,K](f16) @ BT[N,K]^T(f16) + bias(+residual); optional relu.
// All dims exact multiples: M%64==0, N%64==0, K%32==0. Full EXEC everywhere.
// Tiles staged via TDM (tensor_load_to_lds) double-buffered & pipelined.
__global__ __launch_bounds__(128)
void gemm_wmma_kernel(const _Float16* __restrict__ A, int lda,
                      const _Float16* __restrict__ BT, int ldbt,
                      const float* __restrict__ bias, int biasN,
                      const float* __restrict__ residual,
                      float* out32, _Float16* out16, int ldc,
                      int M, int N, int K, int relu) {
  __shared__ alignas(16) _Float16 As [2][64][40];   // 32 cols + 8 pad (16B aligned rows)
  __shared__ alignas(16) _Float16 BTs[2][64][40];

  int tid = threadIdx.x, lane = tid & 31, wid = tid >> 5;
  int bm = blockIdx.y * 64, bn = blockIdx.x * 64;
  int wm = (wid >> 1) * 32, wn = (wid & 1) * 32;
  int ar = lane & 15, kof = (lane >> 4) * 16;
  int bn0 = lane & 15, kb = (lane >> 4) * 16;

  auto stage = [&](int buf, int kt) {
    int ks = kt << 5;
#if HAVE_TDM
    // one 16x32 TDM tile per wave per matrix; LDS rows padded 32->40 halves
    // pad_interval code 3 = 16 DWORDs (64B = 32 halves), pad_amount code 3 = 4 DWORDs (8 halves)
    tdm_load_2d(A  + (size_t)(bm + wid * 16) * lda  + ks, &As [buf][wid * 16][0], 16, 32, lda,  3, 3);
    tdm_load_2d(BT + (size_t)(bn + wid * 16) * ldbt + ks, &BTs[buf][wid * 16][0], 16, 32, ldbt, 3, 3);
#else
#pragma unroll
    for (int c = tid; c < 256; c += 128) {
      int r = c >> 2, ck = (c & 3) * 8;
      *(v8h*)&As[buf][r][ck]  = *(const v8h*)(A  + (size_t)(bm + r) * lda  + ks + ck);
    }
#pragma unroll
    for (int c = tid; c < 256; c += 128) {
      int r = c >> 2, ck = (c & 3) * 8;
      *(v8h*)&BTs[buf][r][ck] = *(const v8h*)(BT + (size_t)(bn + r) * ldbt + ks + ck);
    }
#endif
  };

  v8f acc[2][2] = {};
  int nk = K >> 5;
  stage(0, 0);                                   // prologue
  for (int kt = 0; kt < nk; ++kt) {
    int buf = kt & 1;
    wait_stage();
    __syncthreads();
    if (kt + 1 < nk) stage(buf ^ 1, kt + 1);     // DMA next tile overlaps WMMA

    v16h a[2], bf[2];
#pragma unroll
    for (int t = 0; t < 2; ++t) {
      const v8h* ap = (const v8h*)&As[buf][wm + t * 16 + ar][0];
      a[t] = __builtin_shufflevector(ap[kof >> 3], ap[(kof >> 3) + 1],
                                     0,1,2,3,4,5,6,7,8,9,10,11,12,13,14,15);
      const v8h* bp = (const v8h*)&BTs[buf][wn + t * 16 + bn0][0];
      bf[t] = __builtin_shufflevector(bp[kb >> 3], bp[(kb >> 3) + 1],
                                      0,1,2,3,4,5,6,7,8,9,10,11,12,13,14,15);
    }
#pragma unroll
    for (int tm = 0; tm < 2; ++tm)
#pragma unroll
      for (int tn = 0; tn < 2; ++tn)
        acc[tm][tn] = __builtin_amdgcn_wmma_f32_16x16x32_f16(
            false, a[tm], false, bf[tn], (short)0, acc[tm][tn], false, false);
  }

  int rloc = (lane >> 4) << 3, col = lane & 15;
#pragma unroll
  for (int tm = 0; tm < 2; ++tm)
#pragma unroll
    for (int tn = 0; tn < 2; ++tn)
#pragma unroll
      for (int r = 0; r < 8; ++r) {
        int gr = bm + wm + tm * 16 + rloc + r;
        int gc = bn + wn + tn * 16 + col;
        float v = acc[tm][tn][r] + (gc < biasN ? bias[gc] : 0.f);
        if (relu) v = fmaxf(v, 0.f);
        if (residual) v += residual[(size_t)gr * ldc + gc];
        if (out32) out32[(size_t)gr * ldc + gc] = v;
        if (out16) out16[(size_t)gr * ldc + gc] = (_Float16)v;
      }
}

// ---------------------------------------------------------------- fused attention
// grid = (T/16 row-tiles, B*H). Block = 128 threads (4 waves).
// S = Q Kh^T * scale (WMMA, K tiles TDM-staged) -> f16 in LDS -> softmax
// -> attOut += p/B -> Y = P Vh (WMMA, V tiles TDM-staged).
__global__ __launch_bounds__(128)
void attn_kernel(const _Float16* __restrict__ q16, const _Float16* __restrict__ k16,
                 const _Float16* __restrict__ v16, float* __restrict__ ybuf,
                 float* __restrict__ attOut, float scale) {
  __shared__ alignas(16) _Float16 Qs[16][HDIM];        // 2 KB
  __shared__ alignas(16) _Float16 Ss[16][TSEQ];        // 32 KB (scores -> probs)
  __shared__ alignas(16) _Float16 KVs[2][64][HDIM + 8];// 18 KB double buffer
  __shared__ float red[16][8];

  int rowTile = blockIdx.x, bh = blockIdx.y;
  int b = bh >> 4, h = bh & 15;
  int tid = threadIdx.x, lane = tid & 31, wid = tid >> 5;
  int row0 = rowTile * 16;
  const size_t headOff = (size_t)h * HDIM;
  int ar = lane & 15, kof = (lane >> 4) * 16;
  int n = lane & 15, kb = (lane >> 4) * 16;
  int rbase = (lane >> 4) * 8, col = lane & 15;

  auto stageKV = [&](const _Float16* src, int buf, int it) {
#if HAVE_TDM
    // one 16x64 TDM tile per wave; LDS rows padded 64->72 halves
    // pad_interval code 4 = 32 DWORDs (128B = 64 halves), pad_amount code 3 = 4 DWORDs
    tdm_load_2d(src + ((size_t)b * TSEQ + it * 64 + wid * 16) * CDIM + headOff,
                &KVs[buf][wid * 16][0], 16, 64, CDIM, 4, 3);
#else
#pragma unroll
    for (int c = tid; c < 512; c += 128) {
      int key = c >> 3, ck = (c & 7) * 8;
      *(v8h*)&KVs[buf][key][ck] =
          *(const v8h*)(src + ((size_t)b * TSEQ + it * 64 + key) * CDIM + headOff + ck);
    }
#endif
  };

  // Q tile 16x64 : one 16B chunk per thread
  {
    int r = tid >> 3, ck = (tid & 7) * 8;
    *(v8h*)&Qs[r][ck] = *(const v8h*)(q16 + ((size_t)b * TSEQ + row0 + r) * CDIM + headOff + ck);
  }

  // ---- scores: 16 key-blocks of 64; each wave computes one 16x16 tile per block
  stageKV(k16, 0, 0);
  for (int it = 0; it < 16; ++it) {
    int buf = it & 1;
    wait_stage();
    __syncthreads();
    if (it + 1 < 16) stageKV(k16, buf ^ 1, it + 1);
    v8f acc = {};
#pragma unroll
    for (int ks = 0; ks < 64; ks += 32) {
      const v8h* ap = (const v8h*)&Qs[ar][0];
      v16h a = __builtin_shufflevector(ap[(ks + kof) >> 3], ap[((ks + kof) >> 3) + 1],
                                       0,1,2,3,4,5,6,7,8,9,10,11,12,13,14,15);
      const v8h* bp = (const v8h*)&KVs[buf][wid * 16 + n][0];
      v16h bfr = __builtin_shufflevector(bp[(ks + kb) >> 3], bp[((ks + kb) >> 3) + 1],
                                         0,1,2,3,4,5,6,7,8,9,10,11,12,13,14,15);
      acc = __builtin_amdgcn_wmma_f32_16x16x32_f16(false, a, false, bfr, (short)0,
                                                   acc, false, false);
    }
#pragma unroll
    for (int r = 0; r < 8; ++r)
      Ss[rbase + r][it * 64 + wid * 16 + col] = (_Float16)(acc[r] * scale);
  }
  __syncthreads();

  // ---- softmax: 8 threads per row
  {
    int r = tid >> 3, sub = tid & 7;
    float mx = -1e30f;
    for (int c = sub; c < TSEQ; c += 8) mx = fmaxf(mx, (float)Ss[r][c]);
    red[r][sub] = mx;
    __syncthreads();
    if (sub == 0) { float m = red[r][0]; for (int i = 1; i < 8; ++i) m = fmaxf(m, red[r][i]); red[r][0] = m; }
    __syncthreads();
    float m = red[r][0];
    float s = 0.f;
    for (int c = sub; c < TSEQ; c += 8) s += __expf((float)Ss[r][c] - m);
    __syncthreads();
    red[r][sub] = s;
    __syncthreads();
    if (sub == 0) { float t = 0.f; for (int i = 0; i < 8; ++i) t += red[r][i]; red[r][0] = t; }
    __syncthreads();
    float inv = 1.0f / red[r][0];
    for (int c = sub; c < TSEQ; c += 8) {
      float p = __expf((float)Ss[r][c] - m) * inv;
      Ss[r][c] = (_Float16)p;
      atomicAdd(&attOut[((size_t)h * TSEQ + row0 + r) * TSEQ + c], p * (1.0f / NB));
    }
  }
  __syncthreads();

  // ---- Y = P @ Vh : wave w owns 16-col tile w; accumulate over 16 key blocks
  {
    v8f acc = {};
    stageKV(v16, 0, 0);
    for (int it = 0; it < 16; ++it) {
      int buf = it & 1;
      wait_stage();
      __syncthreads();
      if (it + 1 < 16) stageKV(v16, buf ^ 1, it + 1);
#pragma unroll
      for (int ks = 0; ks < 64; ks += 32) {
        const v8h* ap = (const v8h*)&Ss[ar][it * 64 + ks];
        v16h a = __builtin_shufflevector(ap[kof >> 3], ap[(kof >> 3) + 1],
                                         0,1,2,3,4,5,6,7,8,9,10,11,12,13,14,15);
        v16h bfr;
#pragma unroll
        for (int i = 0; i < 16; ++i) bfr[i] = KVs[buf][ks + kb + i][wid * 16 + n];
        acc = __builtin_amdgcn_wmma_f32_16x16x32_f16(false, a, false, bfr, (short)0,
                                                     acc, false, false);
      }
    }
#pragma unroll
    for (int r = 0; r < 8; ++r)
      ybuf[((size_t)b * TSEQ + row0 + rbase + r) * CDIM + headOff + wid * 16 + col] = acc[r];
  }
}

// ---------------------------------------------------------------- pool
__global__ void pool_kernel(const float* __restrict__ lnf, float* __restrict__ pooled) {
  int i = blockIdx.x * blockDim.x + threadIdx.x;  // B*C
  if (i >= NB * CDIM) return;
  int b = i >> 10, c = i & (CDIM - 1);
  float s = 0.f;
  for (int t = 0; t < TSEQ; ++t) s += lnf[((size_t)b * TSEQ + t) * CDIM + c];
  pooled[i] = s * (1.0f / TSEQ);
}

// ---------------------------------------------------------------- host
extern "C" void kernel_launch(void* const* d_in, const int* in_sizes, int n_in,
                              void* d_out, int out_size, void* d_ws, size_t ws_size,
                              hipStream_t stream) {
  const int*   idx   = (const int*)  d_in[0];
  const float* tok   = (const float*)d_in[1];
  const float* pos   = (const float*)d_in[2];
  const float* Wq    = (const float*)d_in[3];
  const float* bq    = (const float*)d_in[4];
  const float* Wk    = (const float*)d_in[5];
  const float* bk    = (const float*)d_in[6];
  const float* Wv    = (const float*)d_in[7];
  const float* bv    = (const float*)d_in[8];
  const float* Wo    = (const float*)d_in[9];
  const float* bo    = (const float*)d_in[10];
  const float* ln1w  = (const float*)d_in[11];
  const float* ln1b  = (const float*)d_in[12];
  const float* ln2w  = (const float*)d_in[13];
  const float* ln2b  = (const float*)d_in[14];
  const float* W1    = (const float*)d_in[15];
  const float* b1    = (const float*)d_in[16];
  const float* W2    = (const float*)d_in[17];
  const float* b2    = (const float*)d_in[18];
  const float* lnfw  = (const float*)d_in[19];
  const float* lnfb  = (const float*)d_in[20];

  float* pooled = (float*)d_out;
  float* att    = (float*)d_out + (size_t)NB * CDIM;

  const size_t M    = (size_t)NB * TSEQ;        // 4096
  const size_t ACT  = M * CDIM;                 // 4M
  const size_t CC   = (size_t)CDIM * CDIM;
  const size_t LHTT = (size_t)NL * NH * TSEQ * TSEQ;

  char* cur = (char*)d_ws;
  auto carve = [&](size_t bytes) { char* p = cur; cur += (bytes + 255) & ~(size_t)255; return p; };
  float*    X32   = (float*)   carve(ACT * 4);
  float*    Y32   = (float*)   carve(ACT * 4);
  _Float16* H16   = (_Float16*)carve(ACT * 2);
  _Float16* Y16   = (_Float16*)carve(ACT * 2);
  _Float16* Q16   = (_Float16*)carve(ACT * 2);
  _Float16* K16   = (_Float16*)carve(ACT * 2);
  _Float16* V16   = (_Float16*)carve(ACT * 2);
  _Float16* G16   = (_Float16*)carve(M * FFP * 2);
  _Float16* WQT   = (_Float16*)carve(NL * CC * 2);
  _Float16* WKT   = (_Float16*)carve(NL * CC * 2);
  _Float16* WVT   = (_Float16*)carve(NL * CC * 2);
  _Float16* WOT   = (_Float16*)carve(NL * CC * 2);
  _Float16* W1T   = (_Float16*)carve((size_t)NL * FFP * CDIM * 2);  // [FFP][CDIM] per layer
  _Float16* W2T   = (_Float16*)carve((size_t)NL * CDIM * FFP * 2);  // [CDIM][FFP] per layer

  auto blks = [](size_t n) { return (unsigned)((n + 255) / 256); };

  zero4_kernel<<<blks(LHTT / 4), 256, 0, stream>>>(att, LHTT / 4);

  dim3 gT(CDIM / 32, CDIM / 32);
  for (int l = 0; l < NL; ++l) {
    cvtT_kernel<<<gT, 256, 0, stream>>>(Wq + l * CC, WQT + l * CC, CDIM, CDIM, CDIM, CDIM);
    cvtT_kernel<<<gT, 256, 0, stream>>>(Wk + l * CC, WKT + l * CC, CDIM, CDIM, CDIM, CDIM);
    cvtT_kernel<<<gT, 256, 0, stream>>>(Wv + l * CC, WVT + l * CC, CDIM, CDIM, CDIM, CDIM);
    cvtT_kernel<<<gT, 256, 0, stream>>>(Wo + l * CC, WOT + l * CC, CDIM, CDIM, CDIM, CDIM);
    cvtT_kernel<<<dim3(CDIM / 32, FFP / 32), 256, 0, stream>>>(
        W1 + (size_t)l * CDIM * FFD, W1T + (size_t)l * FFP * CDIM, CDIM, FFD, CDIM, FFP);
    cvtT_kernel<<<dim3(FFP / 32, CDIM / 32), 256, 0, stream>>>(
        W2 + (size_t)l * FFD * CDIM, W2T + (size_t)l * CDIM * FFP, FFD, CDIM, FFP, CDIM);
  }

  embed_kernel<<<blks(ACT / 4), 256, 0, stream>>>(idx, tok, pos, X32);

  const float scale = 0.125f;  // 1/sqrt(64)
  dim3 gProj(CDIM / 64, (unsigned)(M / 64));
  dim3 gF1(FFP / 64, (unsigned)(M / 64));

  for (int l = 0; l < NL; ++l) {
    size_t oCC = (size_t)l * CC;
    ln_kernel<<<(unsigned)M, 256, 0, stream>>>(X32, ln1w + (size_t)l * CDIM,
                                               ln1b + (size_t)l * CDIM, H16, nullptr);
    gemm_wmma_kernel<<<gProj, 128, 0, stream>>>(H16, CDIM, WQT + oCC, CDIM,
        bq + (size_t)l * CDIM, CDIM, nullptr, nullptr, Q16, CDIM, (int)M, CDIM, CDIM, 0);
    gemm_wmma_kernel<<<gProj, 128, 0, stream>>>(H16, CDIM, WKT + oCC, CDIM,
        bk + (size_t)l * CDIM, CDIM, nullptr, nullptr, K16, CDIM, (int)M, CDIM, CDIM, 0);
    gemm_wmma_kernel<<<gProj, 128, 0, stream>>>(H16, CDIM, WVT + oCC, CDIM,
        bv + (size_t)l * CDIM, CDIM, nullptr, nullptr, V16, CDIM, (int)M, CDIM, CDIM, 0);
    attn_kernel<<<dim3(TSEQ / 16, NB * NH), 128, 0, stream>>>(
        Q16, K16, V16, Y32, att + (size_t)l * NH * TSEQ * TSEQ, scale);
    cvt4_f32_f16_kernel<<<blks(ACT / 4), 256, 0, stream>>>(Y32, Y16, ACT / 4);
    gemm_wmma_kernel<<<gProj, 128, 0, stream>>>(Y16, CDIM, WOT + oCC, CDIM,
        bo + (size_t)l * CDIM, CDIM, X32, X32, nullptr, CDIM, (int)M, CDIM, CDIM, 0);
    ln_kernel<<<(unsigned)M, 256, 0, stream>>>(X32, ln2w + (size_t)l * CDIM,
                                               ln2b + (size_t)l * CDIM, H16, nullptr);
    gemm_wmma_kernel<<<gF1, 128, 0, stream>>>(H16, CDIM, W1T + (size_t)l * FFP * CDIM, CDIM,
        b1 + (size_t)l * FFD, FFD, nullptr, nullptr, G16, FFP, (int)M, FFP, CDIM, 1);
    gemm_wmma_kernel<<<gProj, 128, 0, stream>>>(G16, FFP, W2T + (size_t)l * CDIM * FFP, FFP,
        b2 + (size_t)l * CDIM, CDIM, X32, X32, nullptr, CDIM, (int)M, CDIM, FFP, 0);
  }

  ln_kernel<<<(unsigned)M, 256, 0, stream>>>(X32, lnfw, lnfb, nullptr, Y32);
  pool_kernel<<<(NB * CDIM + 255) / 256, 256, 0, stream>>>(Y32, pooled);
}
